// SAMI_86414741996164
// MI455X (gfx1250) — compile-verified, matrix operand order checked by reading, not run on hardware
//
#include <hip/hip_runtime.h>
#include <hip/hip_bf16.h>
#include <math.h>

// ---------------- constants (match reference globals) ----------------
constexpr int CB   = 64;     // batch
constexpr int CT   = 196;    // motion tokens
constexpr int CNT  = 77;     // text tokens
constexpr int CH   = 8;      // heads
constexpr int CL   = 64;     // head dim
constexpr int CD   = 512;    // H*L
constexpr int CP   = 256;    // text feat dim
constexpr int CE   = 8;      // experts
constexpr int CNI  = 4;      // intervals
constexpr int CG   = CB / CNI;     // 16 groups
constexpr int CM   = CNI * CL;     // 256  (Mdim)
constexpr int CTE  = 2048;
constexpr int NTOT = CNT + CT;     // 273
constexpr int TOK_TEXT   = CB * CNT;        // 4928
constexpr int TOK_MOTION = CB * CT * CH;    // 100352
constexpr int ROWS_BT    = CB * CT;         // 12544

#define DEV static __device__ __forceinline__

typedef __attribute__((ext_vector_type(16))) _Float16 v16h;
typedef __attribute__((ext_vector_type(8)))  _Float16 h8v;
typedef __attribute__((ext_vector_type(8)))  float    v8f;
typedef __attribute__((ext_vector_type(4)))  float    f4v;
typedef __attribute__((ext_vector_type(4)))  unsigned int u32x4;
typedef __attribute__((ext_vector_type(8)))  int      i32x8;
typedef __attribute__((ext_vector_type(4)))  int      i32x4;

DEV int lane_id() { return (int)(threadIdx.x & 31u); }

DEV float geluf(float x) { return 0.5f * x * (1.0f + erff(x * 0.70710678118654752f)); }
DEV float siluf(float x) { return x / (1.0f + expf(-x)); }

DEV v8f vzero8() {
  v8f z;
#pragma unroll
  for (int i = 0; i < 8; ++i) z[i] = 0.0f;
  return z;
}

// ---- WMMA fragment loaders (layouts per CDNA5 ISA 7.12.2) ----
// A 16x32 f16: lane&15 = M row; lane>>4 picks K halves {kb..kb+7, 16+kb..16+kb+7}.
DEV v16h ldA_lds_f16w(const _Float16* p, int ld) {
  int l = lane_id(); int r = l & 15; int kb = (l >> 4) * 8;
  const _Float16* row = p + (size_t)r * ld + kb;
  h8v lo = *(const h8v*)(row);
  h8v hi = *(const h8v*)(row + 16);
  v16h a;
#pragma unroll
  for (int i = 0; i < 8; ++i) { a[i] = lo[i]; a[8 + i] = hi[i]; }
  return a;
}
// A from row-major f32 global: two pairs of float4 loads + cvt.
DEV v16h ldA_g_f32w(const float* __restrict__ p, int ld) {
  int l = lane_id(); int r = l & 15; int kb = (l >> 4) * 8;
  const float* row = p + (size_t)r * ld + kb;
  f4v a0 = *(const f4v*)(row);
  f4v a1 = *(const f4v*)(row + 4);
  f4v b0 = *(const f4v*)(row + 16);
  f4v b1 = *(const f4v*)(row + 20);
  v16h a;
#pragma unroll
  for (int i = 0; i < 4; ++i) {
    a[i]      = (_Float16)a0[i];
    a[4 + i]  = (_Float16)a1[i];
    a[8 + i]  = (_Float16)b0[i];
    a[12 + i] = (_Float16)b1[i];
  }
  return a;
}
// B fragment from lane-swizzled LDS tile: Bsw[nt][lane][16] halfs, contiguous per lane.
DEV v16h ldB_sw(const _Float16* base) {
  const _Float16* p = base + lane_id() * 16;
  h8v lo = *(const h8v*)(p);
  h8v hi = *(const h8v*)(p + 8);
  v16h b;
#pragma unroll
  for (int i = 0; i < 8; ++i) { b[i] = lo[i]; b[8 + i] = hi[i]; }
  return b;
}
DEV v8f wmma16(v16h a, v16h b, v8f c) {
  return __builtin_amdgcn_wmma_f32_16x16x32_f16(false, a, false, b, (short)0, c, false, false);
}

// ---- stage a 32 x NCOLS K-major tile into swizzled LDS (paired-row b32 stores) ----
// element (k,c) of ntile nt -> lane = ((k>>3)&1)*16 + c, idx i = (k&7) + ((k>>4)&1)*8
union pk2h { _Float16 h[2]; unsigned int u; };
template <int NCOLS, int NTHREADS>
DEV void stage_B_f32(_Float16* Bsw, const float* __restrict__ src, int ld, int tid) {
  // speculative prefetch of the next k-step's slab (slabs advance by 32*ld)
  __builtin_prefetch((const char*)(src + (size_t)32 * ld) + (size_t)tid * 128, 0, 1);
  for (int idx = tid; idx < 16 * NCOLS; idx += NTHREADS) {
    int pr = idx / NCOLS;          // row pair 0..15
    int cc = idx % NCOLS;
    int r = pr * 2;
    int nt = cc >> 4, c = cc & 15;
    int lane = ((r >> 3) & 1) * 16 + c;
    int i0 = (r & 7) + ((r >> 4) & 1) * 8;   // even
    pk2h pk;
    pk.h[0] = (_Float16)src[(size_t)r * ld + cc];
    pk.h[1] = (_Float16)src[(size_t)(r + 1) * ld + cc];
    *(unsigned int*)(Bsw + ((size_t)nt * 32 + lane) * 16 + i0) = pk.u;
  }
}
template <int NCOLS, int NTHREADS>
DEV void stage_B_f16(_Float16* Bsw, const _Float16* __restrict__ src, int ld, int tid) {
  __builtin_prefetch((const char*)(src + (size_t)32 * ld) + (size_t)tid * 128, 0, 1);
  for (int idx = tid; idx < 16 * NCOLS; idx += NTHREADS) {
    int pr = idx / NCOLS;
    int cc = idx % NCOLS;
    int r = pr * 2;
    int nt = cc >> 4, c = cc & 15;
    int lane = ((r >> 3) & 1) * 16 + c;
    int i0 = (r & 7) + ((r >> 4) & 1) * 8;
    pk2h pk;
    pk.h[0] = src[(size_t)r * ld + cc];
    pk.h[1] = src[(size_t)(r + 1) * ld + cc];
    *(unsigned int*)(Bsw + ((size_t)nt * 32 + lane) * 16 + i0) = pk.u;
  }
}

// ---- TDM: async 2D f16 tile load Global->LDS (CDNA5 ISA D#) ----
// tile_x elems contiguous (data_size=2B), tile_y rows, row stride in elems.
DEV void tdm_load_2d_f16(const void* gptr, unsigned lds_off,
                         int tile_x, int tile_y, int row_stride) {
  unsigned long long ga = (unsigned long long)gptr;
  u32x4 g0;
  g0[0] = 1u;                                        // count=1, user descriptor
  g0[1] = lds_off;                                   // lds_addr (bytes)
  g0[2] = (unsigned)(ga & 0xffffffffu);              // global_addr[31:0]
  g0[3] = (unsigned)((ga >> 32) & 0x01ffffffu) | (2u << 30);  // addr[56:32] | type=2
  unsigned td0 = 1u << 20, td1 = 1u << 20;           // huge tensor dims: no OOB clip
  unsigned long long s0 = (unsigned long long)(unsigned)row_stride;
  i32x8 g1;
  g1[0] = (int)(1u << 16);                                        // data_size=1 (2B)
  g1[1] = (int)((td0 & 0xffffu) << 16);                           // tensor_dim0 lo
  g1[2] = (int)((td0 >> 16) | ((td1 & 0xffffu) << 16));           // td0 hi | td1 lo
  g1[3] = (int)((td1 >> 16) | ((unsigned)tile_x << 16));          // td1 hi | tile_dim0
  g1[4] = (int)(unsigned)tile_y;                                  // tile_dim1 (tile_dim2=0)
  g1[5] = (int)(unsigned)(s0 & 0xffffffffu);                      // dim0 stride lo
  g1[6] = (int)(unsigned)((s0 >> 32) & 0xffffu);                  // stride hi | dim1 stride=0
  g1[7] = 0;
  i32x4 z4;
#pragma unroll
  for (int i = 0; i < 4; ++i) z4[i] = 0;
#if __clang_major__ >= 23
  i32x8 z8;
#pragma unroll
  for (int i = 0; i < 8; ++i) z8[i] = 0;
  __builtin_amdgcn_tensor_load_to_lds(g0, g1, z4, z4, z8, 0);
#else
  __builtin_amdgcn_tensor_load_to_lds(g0, g1, z4, z4, 0);
#endif
}

// ================= LayerNorm (+positional embedding) =================
__global__ void ln_text_kernel(const float* __restrict__ xf, const float* __restrict__ g,
                               const float* __restrict__ b, const float* __restrict__ pe,
                               float* __restrict__ out) {
  int tok = blockIdx.x * blockDim.x + threadIdx.x;
  if (tok >= TOK_TEXT) return;
  int n = tok % CNT;
  const float* r = xf + (size_t)tok * CP;
  float m = 0.f;
  for (int c = 0; c < CP; ++c) m += r[c];
  m *= (1.0f / CP);
  float v = 0.f;
  for (int c = 0; c < CP; ++c) { float d = r[c] - m; v += d * d; }
  v *= (1.0f / CP);
  float rs = rsqrtf(v + 1e-5f);
  float* o = out + (size_t)tok * CP;
  for (int c = 0; c < CP; ++c)
    o[c] = (r[c] - m) * rs * g[c] + b[c] + pe[(size_t)n * CP + c];
}

__global__ void ln_motion_kernel(const float* __restrict__ x, const float* __restrict__ g,
                                 const float* __restrict__ b, const float* __restrict__ pe,
                                 float* __restrict__ out) {
  int tok = blockIdx.x * blockDim.x + threadIdx.x;   // ((b*T)+t)*H + h
  if (tok >= TOK_MOTION) return;
  int h = tok & (CH - 1);
  int bt = tok >> 3;                 // b*T + t
  int t = bt % CT;
  const float* r = x + (size_t)bt * CD + h * CL;
  float m = 0.f;
  for (int c = 0; c < CL; ++c) m += r[c];
  m *= (1.0f / CL);
  float v = 0.f;
  for (int c = 0; c < CL; ++c) { float d = r[c] - m; v += d * d; }
  v *= (1.0f / CL);
  float rs = rsqrtf(v + 1e-5f);
  float* o = out + (size_t)tok * CL;
  const float* p = pe + ((size_t)t * CH + h) * CL;
  for (int c = 0; c < CL; ++c)
    o[c] = (r[c] - m) * rs * g[c] + b[c] + p[c];
}

// ================= MoE gating: softmax + top-2 -> comb =================
__global__ void gate_kernel(const float* __restrict__ Xln, const float* __restrict__ gw,
                            float* __restrict__ comb, int Ntok, int DIN) {
  int tok = blockIdx.x * blockDim.x + threadIdx.x;
  if (tok >= Ntok) return;
  float s[CE];
#pragma unroll
  for (int e = 0; e < CE; ++e) s[e] = 0.f;
  const float* r = Xln + (size_t)tok * DIN;
  for (int c = 0; c < DIN; ++c) {
    float xv = r[c];
    const float* gr = gw + (size_t)c * CE;
#pragma unroll
    for (int e = 0; e < CE; ++e) s[e] += xv * gr[e];
  }
  float mx = s[0];
#pragma unroll
  for (int e = 1; e < CE; ++e) mx = fmaxf(mx, s[e]);
  float sum = 0.f;
#pragma unroll
  for (int e = 0; e < CE; ++e) { s[e] = expf(s[e] - mx); sum += s[e]; }
  float inv = 1.0f / sum;
#pragma unroll
  for (int e = 0; e < CE; ++e) s[e] *= inv;
  int i1 = 0;
#pragma unroll
  for (int e = 1; e < CE; ++e) if (s[e] > s[i1]) i1 = e;
  int i2 = (i1 == 0) ? 1 : 0;
#pragma unroll
  for (int e = 0; e < CE; ++e) if (e != i1 && s[e] > s[i2]) i2 = e;
  float* o = comb + (size_t)tok * CE;
#pragma unroll
  for (int e = 0; e < CE; ++e) o[e] = (e == i1 || e == i2) ? s[e] : 0.0f;
}

// ============== Fused MoE expert FFN (WMMA), writes gelu(y) ==============
template <int DIN, int DHID, int TOKS>
__global__ void moe_kernel(const float* __restrict__ Xln, const float* __restrict__ comb,
                           const float* __restrict__ w1, const float* __restrict__ b1,
                           const float* __restrict__ w2, const float* __restrict__ b2,
                           float* __restrict__ Y) {
  constexpr int MT  = TOKS / 16;              // m-tiles (1 or 2)
  constexpr int WPM = 8 / MT;                 // waves per m-tile (8 or 4)
  constexpr int NCH = DHID / 256;             // up n-chunks of 16 ntiles (4 or 1)
  constexpr int NUW = 16 / WPM;               // up ntiles per wave per chunk (2 or 4)
  constexpr int NDW = (DIN / 16) / WPM;       // down ntiles per wave (2 or 1)
  __shared__ alignas(16) _Float16 Hs[TOKS][DHID];
  __shared__ alignas(16) _Float16 Bsw[16][32][16];
  const int tid = (int)threadIdx.x;
  const int wave = tid >> 5;
  const int lane = lane_id();
  const int tok0 = blockIdx.x * TOKS;
  const int mt = wave / WPM;
  const int wj = wave % WPM;
  v8f yacc[NDW];
#pragma unroll
  for (int q = 0; q < NDW; ++q) yacc[q] = vzero8();

  for (int e = 0; e < CE; ++e) {
    const float* w1e = w1 + (size_t)e * DIN * DHID;
    const float* w2e = w2 + (size_t)e * DHID * DIN;
    const float* b1e = b1 + (size_t)e * DHID;
    const float* b2e = b2 + (size_t)e * DIN;
    // ---- up: Hs = gelu(X @ w1e + b1e), f16 in LDS ----
    for (int ch = 0; ch < NCH; ++ch) {
      v8f acc[NUW];
#pragma unroll
      for (int q = 0; q < NUW; ++q) acc[q] = vzero8();
#pragma unroll 2
      for (int k = 0; k < DIN / 32; ++k) {
        stage_B_f32<256, 256>(&Bsw[0][0][0], w1e + (size_t)(k * 32) * DHID + ch * 256, DHID, tid);
        __syncthreads();
        v16h a = ldA_g_f32w(Xln + (size_t)(tok0 + mt * 16) * DIN + k * 32, DIN);
#pragma unroll
        for (int q = 0; q < NUW; ++q) {
          v16h bf = ldB_sw(&Bsw[wj * NUW + q][0][0]);
          acc[q] = wmma16(a, bf, acc[q]);
        }
        __syncthreads();
      }
      int rb = mt * 16 + (lane >> 4) * 8;
#pragma unroll
      for (int q = 0; q < NUW; ++q) {
        int col = ch * 256 + (wj * NUW + q) * 16 + (lane & 15);
        float bb = b1e[col];
#pragma unroll
        for (int v = 0; v < 8; ++v)
          Hs[rb + v][col] = (_Float16)geluf(acc[q][v] + bb);
      }
    }
    // ---- down: yacc += comb * (Hs @ w2e + b2e) ----
    v8f dacc[NDW];
#pragma unroll
    for (int q = 0; q < NDW; ++q) dacc[q] = vzero8();
#pragma unroll 2
    for (int k = 0; k < DHID / 32; ++k) {
      stage_B_f32<DIN, 256>(&Bsw[0][0][0], w2e + (size_t)(k * 32) * DIN, DIN, tid);
      __syncthreads();   // also makes Hs stores visible before reads
      v16h a = ldA_lds_f16w(&Hs[mt * 16][k * 32], DHID);
#pragma unroll
      for (int q = 0; q < NDW; ++q) {
        v16h bf = ldB_sw(&Bsw[wj * NDW + q][0][0]);
        dacc[q] = wmma16(a, bf, dacc[q]);
      }
      __syncthreads();
    }
    int rb = mt * 16 + (lane >> 4) * 8;
#pragma unroll
    for (int q = 0; q < NDW; ++q) {
      int col = (wj * NDW + q) * 16 + (lane & 15);
      float bb = b2e[col];
#pragma unroll
      for (int v = 0; v < 8; ++v) {
        int row = tok0 + rb + v;
        float cmb = comb[(size_t)row * CE + e];
        yacc[q][v] += cmb * (dacc[q][v] + bb);
      }
    }
  }
  // ---- write gelu(y) ----
  int rb = mt * 16 + (lane >> 4) * 8;
#pragma unroll
  for (int q = 0; q < NDW; ++q) {
    int col = (wj * NDW + q) * 16 + (lane & 15);
#pragma unroll
    for (int v = 0; v < 8; ++v)
      Y[(size_t)(tok0 + rb + v) * DIN + col] = geluf(yacc[q][v]);
  }
}

// ================= MoE output projection: feat = Yg @ pw + pb =================
__global__ void proj_kernel(const float* __restrict__ Yg, const float* __restrict__ pw,
                            const float* __restrict__ pb, float* __restrict__ feat,
                            int Ntok, int DIN, int DOUT) {
  long long idx = (long long)blockIdx.x * blockDim.x + threadIdx.x;
  if (idx >= (long long)Ntok * DOUT) return;
  int o = (int)(idx % DOUT);
  long long tok = idx / DOUT;
  float s = pb[o];
  const float* r = Yg + (size_t)tok * DIN;
  for (int c = 0; c < DIN; ++c) s += r[c] * pw[(size_t)c * DOUT + o];
  feat[idx] = s;
}

// ================= body_feat -> hbuf =================
__global__ void body_kernel(const float* __restrict__ mf, const float* __restrict__ bw,
                            float* __restrict__ hbuf) {
  long long idx = (long long)blockIdx.x * blockDim.x + threadIdx.x;
  if (idx >= (long long)CB * CT * CD) return;
  int dc = (int)(idx & (CD - 1));
  int h = dc >> 6, d = dc & 63;
  long long bt = idx >> 9;
  float w[CH]; float mx = -1e30f;
#pragma unroll
  for (int l = 0; l < CH; ++l) { w[l] = bw[h * CH + l]; mx = fmaxf(mx, w[l]); }
  float s = 0.f;
#pragma unroll
  for (int l = 0; l < CH; ++l) { w[l] = expf(w[l] - mx); s += w[l]; }
  float inv = 1.0f / s, acc = 0.f;
#pragma unroll
  for (int l = 0; l < CH; ++l)
    acc += w[l] * inv * mf[((size_t)bt * CH + l) * (3 * CL) + d];
  hbuf[idx] = acc;
}

// ======= key-softmax (over N=273) + template = K^T V  (per b,h) =======
__global__ void template_kernel(const float* __restrict__ tf,  // (B,77,128)
                                const float* __restrict__ mf,  // (B,196,8,192)
                                const float* __restrict__ srcm,
                                const int* __restrict__ cond,
                                float* __restrict__ tmpl) {    // (B,8,64,64)
  const int h = blockIdx.x, b = blockIdx.y;
  const int d = (int)threadIdx.x;                              // 0..63
  const float tct = ((cond[b] % 10) > 0) ? 1.0f : 0.0f;
  const float kb_t = (1.0f - tct) * (-1e6f);
  float mx = -1e30f;
  for (int n = 0; n < CNT; ++n)
    mx = fmaxf(mx, tf[((size_t)b * CNT + n) * (2 * CL) + d] + kb_t);
  for (int t = 0; t < CT; ++t) {
    float sm = srcm[b * CT + t];
    float kv = mf[(((size_t)b * CT + t) * CH + h) * (3 * CL) + CL + d] + (1.0f - sm) * (-1e6f);
    mx = fmaxf(mx, kv);
  }
  float sum = 0.f;
  for (int n = 0; n < CNT; ++n)
    sum += expf(tf[((size_t)b * CNT + n) * (2 * CL) + d] + kb_t - mx);
  for (int t = 0; t < CT; ++t) {
    float sm = srcm[b * CT + t];
    sum += expf(mf[(((size_t)b * CT + t) * CH + h) * (3 * CL) + CL + d] + (1.0f - sm) * (-1e6f) - mx);
  }
  const float inv = 1.0f / sum;
  float acc[CL];
#pragma unroll
  for (int l = 0; l < CL; ++l) acc[l] = 0.f;
  __shared__ float vrow[16][CL];
  for (int c0 = 0; c0 < NTOT; c0 += 16) {
    __syncthreads();
    for (int idx = (int)threadIdx.x; idx < 16 * CL; idx += 64) {
      int rr = idx >> 6, cc = idx & 63;
      int n = c0 + rr;
      float val = 0.f;
      if (n < NTOT) {
        if (n < CNT) val = tf[((size_t)b * CNT + n) * (2 * CL) + CL + cc] * tct;
        else {
          int t = n - CNT;
          val = mf[(((size_t)b * CT + t) * CH + h) * (3 * CL) + 2 * CL + cc] * srcm[b * CT + t];
        }
      }
      vrow[rr][cc] = val;
    }
    __syncthreads();
    for (int rr = 0; rr < 16 && c0 + rr < NTOT; ++rr) {
      int n = c0 + rr;
      float kv;
      if (n < CNT) kv = tf[((size_t)b * CNT + n) * (2 * CL) + d] + kb_t;
      else {
        int t = n - CNT;
        kv = mf[(((size_t)b * CT + t) * CH + h) * (3 * CL) + CL + d] + (1.0f - srcm[b * CT + t]) * (-1e6f);
      }
      float e = expf(kv - mx) * inv;
#pragma unroll
      for (int l = 0; l < CL; ++l) acc[l] += e * vrow[rr][l];
    }
  }
  float* o = tmpl + (((size_t)b * CH + h) * CL + d) * CL;
#pragma unroll
  for (int l = 0; l < CL; ++l) o[l] = acc[l];
}

// ================= tt = sigmoid(ffn(template)/t_sigma) * ml/196 =================
__global__ void tt_kernel(const float* __restrict__ tmpl, const float* __restrict__ w1,
                          const float* __restrict__ b1, const float* __restrict__ w2,
                          const float* __restrict__ b2, const float* __restrict__ tsig,
                          const int* __restrict__ mlen, float* __restrict__ tt) {
  int row = blockIdx.x * 8 + (int)(threadIdx.x >> 5);   // (b*8+h)*64+d, 0..32767
  int lane = lane_id();
  int b = row >> 9;
  const float* tr = tmpl + (size_t)row * CL;
  float partial = 0.f;
#pragma unroll
  for (int jj = 0; jj < 8; ++jj) {
    int j = lane * 8 + jj;
    float s = b1[j];
    for (int q = 0; q < CL; ++q) s += tr[q] * w1[(size_t)q * 256 + j];
    partial += geluf(s) * w2[j];
  }
  for (int off = 16; off > 0; off >>= 1) partial += __shfl_down(partial, off, 32);
  if (lane == 0) {
    float v = partial + b2[0];
    float sg = 1.0f / (1.0f + expf(-v / tsig[0]));
    tt[row] = sg * (float)mlen[b] * (1.0f / (float)CT);
  }
}

// ===== template FFNs (k=0..3) + cubic coeffs -> AcatH f16 (B operand of mix) =====
__global__ void tmplffn_kernel(const float* __restrict__ tmpl, const float* __restrict__ w1,
                               const float* __restrict__ b1, const float* __restrict__ w2,
                               const float* __restrict__ b2, const float* __restrict__ tt,
                               const int* __restrict__ mlen, _Float16* __restrict__ AcatH) {
  int r = blockIdx.x;                 // (g*8+h)*256 + m, 0..32767
  int g = r >> 11;
  int h = (r >> 8) & 7;
  int m = r & 255;
  int i = m >> 6, d = m & 63;
  int b = g * CNI + i;
  int l = (int)threadIdx.x;           // 0..63
  __shared__ float row_s[CL];
  __shared__ float hid[256];
  row_s[l] = tmpl[(((size_t)b * CH + h) * CL + d) * CL + l];
  __syncthreads();
  float f[4];
#pragma unroll 1
  for (int k = 0; k < 4; ++k) {
    const float* w1k = w1 + (size_t)k * CL * 256;
    const float* b1k = b1 + (size_t)k * 256;
    const float* w2k = w2 + (size_t)k * 256 * CL;
    const float* b2k = b2 + (size_t)k * CL;
#pragma unroll
    for (int jj = 0; jj < 4; ++jj) {
      int j = l * 4 + jj;
      float s = b1k[j];
      for (int q = 0; q < CL; ++q) s += row_s[q] * w1k[(size_t)q * 256 + j];
      hid[j] = geluf(s);
    }
    __syncthreads();
    float o = b2k[l];
    for (int j = 0; j < 256; ++j) o += hid[j] * w2k[(size_t)j * CL + l];
    f[k] = o;
    __syncthreads();
  }
  float offs = 0.f;
  for (int q = 0; q < i; ++q) offs += (float)mlen[g * CNI + q];
  offs *= (1.0f / (float)CT);
  float tc = tt[((size_t)b * CH + h) * CL + d] + offs;
  float tc2 = tc * tc, tc3 = tc2 * tc;
  float a0 = f[0] - f[1] * tc + f[2] * tc2 - f[3] * tc3;
  float a1 = f[1] - 2.0f * f[2] * tc + 3.0f * f[3] * tc2;
  float a2 = f[2] - 3.0f * f[3] * tc;
  size_t base = (size_t)r * 256;
  AcatH[base + l]         = (_Float16)a0;
  AcatH[base + 64 + l]    = (_Float16)a1;
  AcatH[base + 128 + l]   = (_Float16)a2;
  AcatH[base + 192 + l]   = (_Float16)f[3];
}

// ====== time_coef softmax + mix GEMM (WMMA) + Horner combine; hbuf += y_t ======
__global__ void mix_kernel(const _Float16* __restrict__ AcatH, const float* __restrict__ tt,
                           const int* __restrict__ mlen, const float* __restrict__ sigma,
                           float* __restrict__ hbuf) {
  const int bx = blockIdx.x;               // tau tile (32 rows), 0..6
  const int h  = blockIdx.y;
  const int b  = blockIdx.z;
  const int g = b >> 2, iv = b & 3;
  __shared__ alignas(16) _Float16 tco[32][256];
  __shared__ alignas(16) _Float16 slab[16][32][16];   // swizzled B tiles
  __shared__ float ttm[256];
  __shared__ float ts_[32];
  const int tid = (int)threadIdx.x, lane = lane_id(), wave = tid >> 5;
  float offs4[4];
  {
    float c = 0.f;
#pragma unroll
    for (int q = 0; q < 4; ++q) { offs4[q] = c * (1.0f / (float)CT); c += (float)mlen[g * CNI + q]; }
  }
  if (tid < 256) {
    int m = tid, im = m >> 6, d = m & 63, bb = g * CNI + im;
    ttm[m] = tt[((size_t)bb * CH + h) * CL + d] + offs4[im];
  }
  if (tid < 32) ts_[tid] = (float)(bx * 32 + tid) * (1.0f / (float)CT) + offs4[iv];
  __syncthreads();
  const float sg = sigma[0];
  if (tid < 32) {
    int tau = bx * 32 + tid;
    if (tau < CT) {
      float tv = ts_[tid], mx = -1e30f;
      for (int m = 0; m < 256; ++m) {
        float dd = (tv - ttm[m]) * (float)CT;
        mx = fmaxf(mx, -(dd * dd) / sg);
      }
      float s = 0.f;
      for (int m = 0; m < 256; ++m) {
        float dd = (tv - ttm[m]) * (float)CT;
        float e = expf(-(dd * dd) / sg - mx);
        s += e;
        tco[tid][m] = (_Float16)e;
      }
      float inv = 1.0f / s;
      for (int m = 0; m < 256; ++m) tco[tid][m] = (_Float16)((float)tco[tid][m] * inv);
    } else {
      for (int m = 0; m < 256; ++m) tco[tid][m] = (_Float16)0.0f;
    }
  }
  __syncthreads();
  const int mt = wave >> 2, j = wave & 3;   // wave owns cols l=16j..16j+15 for a0,a1,a2,tj
  v8f c4[4];
#pragma unroll
  for (int q = 0; q < 4; ++q) c4[q] = vzero8();
  const _Float16* Abase = AcatH + (size_t)(g * CH + h) * 256 * 256;
#pragma unroll 2
  for (int k = 0; k < 8; ++k) {
    stage_B_f16<256, 256>(&slab[0][0][0], Abase + (size_t)(k * 32) * 256, 256, tid);
    __syncthreads();
    v16h a = ldA_lds_f16w(&tco[mt * 16][k * 32], 256);
#pragma unroll
    for (int q = 0; q < 4; ++q) {
      v16h bf = ldB_sw(&slab[j + 4 * q][0][0]);
      c4[q] = wmma16(a, bf, c4[q]);
    }
    __syncthreads();
  }
  const int l = j * 16 + (lane & 15);
  const int rb = mt * 16 + (lane >> 4) * 8;
#pragma unroll
  for (int v = 0; v < 8; ++v) {
    int rloc = rb + v, tau = bx * 32 + rloc;
    if (tau < CT) {
      float tb = ts_[rloc];
      float y = c4[0][v] + tb * (c4[1][v] + tb * (c4[2][v] + tb * c4[3][v]));
      size_t o = (((size_t)b * CT + tau) * CH + h) * CL + l;
      hbuf[o] += y;
    }
  }
}

// ================= eo = silu(emb) @ se_w + se_b =================
__global__ void eo_kernel(const float* __restrict__ emb, const float* __restrict__ sw,
                          const float* __restrict__ sb, float* __restrict__ eo) {
  int idx = blockIdx.x * blockDim.x + threadIdx.x;   // 64*1024
  if (idx >= CB * 2 * CD) return;
  int b = idx >> 10, o = idx & 1023;
  float s = sb[o];
  const float* er = emb + (size_t)b * CTE;
  for (int k = 0; k < CTE; ++k) s += siluf(er[k]) * sw[(size_t)k * (2 * CD) + o];
  eo[idx] = s;
}

// ======== hn: LN(h)*(1+eo_lo)+eo_hi, then silu -> f16 for final GEMM ========
__global__ void hn_kernel(const float* __restrict__ hbuf, const float* __restrict__ g,
                          const float* __restrict__ bb, const float* __restrict__ eo,
                          _Float16* __restrict__ hnH) {
  int tok = blockIdx.x * blockDim.x + threadIdx.x;
  if (tok >= ROWS_BT) return;
  int b = tok / CT;
  const float* r = hbuf + (size_t)tok * CD;
  float m = 0.f;
  for (int c = 0; c < CD; ++c) m += r[c];
  m *= (1.0f / CD);
  float v = 0.f;
  for (int c = 0; c < CD; ++c) { float d = r[c] - m; v += d * d; }
  v *= (1.0f / CD);
  float rs = rsqrtf(v + 1e-5f);
  const float* e0 = eo + (size_t)b * (2 * CD);
  _Float16* o = hnH + (size_t)tok * CD;
  for (int c = 0; c < CD; ++c) {
    float ln = (r[c] - m) * rs * g[c] + bb[c];
    float hv = ln * (1.0f + e0[c]) + e0[CD + c];
    o[c] = (_Float16)siluf(hv);
  }
}

// ========== final: out = x + silu(hn) @ so_w + so_b  (WMMA GEMM) ==========
// A tile is DMA'd by the Tensor Data Mover; B tile staged + swizzled by threads.
__global__ void out_gemm_kernel(const _Float16* __restrict__ hnH, const float* __restrict__ so_w,
                                const float* __restrict__ so_b, const float* __restrict__ x,
                                float* __restrict__ out) {
  __shared__ alignas(16) _Float16 As[64][32];
  __shared__ alignas(16) _Float16 Bs[4][32][16];   // swizzled
  const int tid = (int)threadIdx.x, lane = lane_id(), wave = tid >> 5;
  const int r0 = blockIdx.x * 64, c0 = blockIdx.y * 64;
  const int mt = wave >> 1, ntb = (wave & 1) * 2;
  const unsigned as_lds = (unsigned)(unsigned long long)&As[0][0];
  v8f c2[2];
  c2[0] = vzero8(); c2[1] = vzero8();
#pragma unroll 2
  for (int k = 0; k < 16; ++k) {
    if (wave == 0)   // one TDM issue per block: 64x32 f16 tile, row stride CD
      tdm_load_2d_f16(hnH + (size_t)r0 * CD + k * 32, as_lds, 32, 64, CD);
    stage_B_f32<64, 256>(&Bs[0][0][0], so_w + (size_t)(k * 32) * CD + c0, CD, tid);
    if (wave == 0) __builtin_amdgcn_s_wait_tensorcnt((short)0);
    __syncthreads();
    v16h a = ldA_lds_f16w(&As[mt * 16][0], 32);
#pragma unroll
    for (int q = 0; q < 2; ++q) {
      v16h bf = ldB_sw(&Bs[ntb + q][0][0]);
      c2[q] = wmma16(a, bf, c2[q]);
    }
    __syncthreads();
  }
  const int rb = mt * 16 + (lane >> 4) * 8;
#pragma unroll
  for (int q = 0; q < 2; ++q) {
    int col = c0 + (ntb + q) * 16 + (lane & 15);
    float sb = so_b[col];
#pragma unroll
    for (int v = 0; v < 8; ++v) {
      size_t row = (size_t)(r0 + rb + v);
      out[row * CD + col] = x[row * CD + col] + sb + c2[q][v];
    }
  }
}

// ============================= host launcher =============================
extern "C" void kernel_launch(void* const* d_in, const int* in_sizes, int n_in,
                              void* d_out, int out_size, void* d_ws, size_t ws_size,
                              hipStream_t stream) {
  (void)in_sizes; (void)n_in; (void)out_size; (void)ws_size;
  const float* x        = (const float*)d_in[0];
  const float* xf       = (const float*)d_in[1];
  const float* emb      = (const float*)d_in[2];
  const float* sigma    = (const float*)d_in[3];
  const float* t_sigma  = (const float*)d_in[4];
  const float* norm_g   = (const float*)d_in[5];
  const float* norm_b   = (const float*)d_in[6];
  const float* tnorm_g  = (const float*)d_in[7];
  const float* tnorm_b  = (const float*)d_in[8];
  const float* body_w   = (const float*)d_in[9];
  const float* tm_pe    = (const float*)d_in[10];
  const float* tm_gate  = (const float*)d_in[11];
  const float* tm_w1    = (const float*)d_in[12];
  const float* tm_b1    = (const float*)d_in[13];
  const float* tm_w2    = (const float*)d_in[14];
  const float* tm_b2    = (const float*)d_in[15];
  const float* tm_pw    = (const float*)d_in[16];
  const float* tm_pb    = (const float*)d_in[17];
  const float* mm_pe    = (const float*)d_in[18];
  const float* mm_gate  = (const float*)d_in[19];
  const float* mm_w1    = (const float*)d_in[20];
  const float* mm_b1    = (const float*)d_in[21];
  const float* mm_w2    = (const float*)d_in[22];
  const float* mm_b2    = (const float*)d_in[23];
  const float* mm_pw    = (const float*)d_in[24];
  const float* mm_pb    = (const float*)d_in[25];
  const float* tmpl_w1  = (const float*)d_in[26];
  const float* tmpl_b1  = (const float*)d_in[27];
  const float* tmpl_w2  = (const float*)d_in[28];
  const float* tmpl_b2  = (const float*)d_in[29];
  const float* tt_w1    = (const float*)d_in[30];
  const float* tt_b1    = (const float*)d_in[31];
  const float* tt_w2    = (const float*)d_in[32];
  const float* tt_b2    = (const float*)d_in[33];
  const float* sb_g     = (const float*)d_in[34];
  const float* sb_b     = (const float*)d_in[35];
  const float* se_w     = (const float*)d_in[36];
  const float* se_b     = (const float*)d_in[37];
  const float* so_w     = (const float*)d_in[38];
  const float* so_b     = (const float*)d_in[39];
  const float* src_mask = (const float*)d_in[40];
  const int*   cond     = (const int*)d_in[41];
  const int*   mlen     = (const int*)d_in[42];
  float* out = (float*)d_out;

  // workspace carve-up
  char* W = (char*)d_ws;
  size_t off = 0;
  auto alloc = [&](size_t bytes) { size_t p = off; off = (off + bytes + 255) & ~(size_t)255; return p; };
  float*     xt_text   = (float*)(W + alloc((size_t)TOK_TEXT * CP * 4));
  float*     comb_text = (float*)(W + alloc((size_t)TOK_TEXT * CE * 4));
  float*     y_text    = (float*)(W + alloc((size_t)TOK_TEXT * CP * 4));
  float*     text_feat = (float*)(W + alloc((size_t)TOK_TEXT * 2 * CL * 4));
  float*     xm_ln     = (float*)(W + alloc((size_t)TOK_MOTION * CL * 4));
  float*     comb_m    = (float*)(W + alloc((size_t)TOK_MOTION * CE * 4));
  float*     y_m       = (float*)(W + alloc((size_t)TOK_MOTION * CL * 4));
  float*     mot_feat  = (float*)(W + alloc((size_t)TOK_MOTION * 3 * CL * 4));
  float*     hbuf      = (float*)(W + alloc((size_t)ROWS_BT * CD * 4));
  float*     tmplb     = (float*)(W + alloc((size_t)CB * CH * CL * CL * 4));
  float*     ttb       = (float*)(W + alloc((size_t)CB * CH * CL * 4));
  _Float16*  AcatH     = (_Float16*)(W + alloc((size_t)CG * CH * CM * 256 * 2));
  float*     eo        = (float*)(W + alloc((size_t)CB * 2 * CD * 4));
  _Float16*  hnH       = (_Float16*)(W + alloc((size_t)ROWS_BT * CD * 2));

  // ---- text branch ----
  ln_text_kernel<<<(TOK_TEXT + 255) / 256, 256, 0, stream>>>(xf, tnorm_g, tnorm_b, tm_pe, xt_text);
  gate_kernel<<<(TOK_TEXT + 255) / 256, 256, 0, stream>>>(xt_text, tm_gate, comb_text, TOK_TEXT, CP);
  moe_kernel<CP, 4 * CP, 16><<<TOK_TEXT / 16, 256, 0, stream>>>(xt_text, comb_text, tm_w1, tm_b1, tm_w2, tm_b2, y_text);
  proj_kernel<<<((long long)TOK_TEXT * 2 * CL + 255) / 256, 256, 0, stream>>>(y_text, tm_pw, tm_pb, text_feat, TOK_TEXT, CP, 2 * CL);

  // ---- motion branch ----
  ln_motion_kernel<<<(TOK_MOTION + 255) / 256, 256, 0, stream>>>(x, norm_g, norm_b, mm_pe, xm_ln);
  gate_kernel<<<(TOK_MOTION + 255) / 256, 256, 0, stream>>>(xm_ln, mm_gate, comb_m, TOK_MOTION, CL);
  moe_kernel<CL, 4 * CL, 32><<<TOK_MOTION / 32, 256, 0, stream>>>(xm_ln, comb_m, mm_w1, mm_b1, mm_w2, mm_b2, y_m);
  proj_kernel<<<((long long)TOK_MOTION * 3 * CL + 255) / 256, 256, 0, stream>>>(y_m, mm_pw, mm_pb, mot_feat, TOK_MOTION, CL, 3 * CL);

  // ---- body feat into hbuf ----
  body_kernel<<<((long long)CB * CT * CD + 255) / 256, 256, 0, stream>>>(mot_feat, body_w, hbuf);

  // ---- attention template + tt + cubic coefficients ----
  template_kernel<<<dim3(CH, CB), 64, 0, stream>>>(text_feat, mot_feat, src_mask, cond, tmplb);
  tt_kernel<<<(CB * CH * CL) / 8, 256, 0, stream>>>(tmplb, tt_w1, tt_b1, tt_w2, tt_b2, t_sigma, mlen, ttb);
  tmplffn_kernel<<<CG * CH * CM, 64, 0, stream>>>(tmplb, tmpl_w1, tmpl_b1, tmpl_w2, tmpl_b2, ttb, mlen, AcatH);

  // ---- time_coef + mix (hbuf += y_t) ----
  mix_kernel<<<dim3((CT + 31) / 32, CH, CB), 256, 0, stream>>>(AcatH, ttb, mlen, sigma, hbuf);

  // ---- modulation + output projection ----
  eo_kernel<<<(CB * 2 * CD + 255) / 256, 256, 0, stream>>>(emb, se_w, se_b, eo);
  hn_kernel<<<(ROWS_BT + 255) / 256, 256, 0, stream>>>(hbuf, sb_g, sb_b, eo, hnH);
  out_gemm_kernel<<<dim3(ROWS_BT / 64, CD / 64), 256, 0, stream>>>(hnH, so_w, so_b, x, out);
}
